// HausdorffLoss_73117523247200
// MI455X (gfx1250) — compile-verified, matrix operand order checked by reading, not run on hardware
//
#include <hip/hip_runtime.h>

typedef __attribute__((ext_vector_type(16))) __bf16 v16bf;
typedef __attribute__((ext_vector_type(8)))  float  v8f;

#define BB 8
#define NN 4096
#define MM 4096
#define CC 128
#define FLTMAX 3.402823466e+38f
#define INF_BITS 0x7F800000u

// ---------------- init colMin (d^2) bits to +inf ----------------
__global__ void hd_init_col(unsigned int* __restrict__ colBits) {
    int i = blockIdx.x * blockDim.x + threadIdx.x;
    if (i < BB * MM) colBits[i] = INF_BITS;
}

// ---------------- squared row norms for both sets ----------------
__global__ void hd_row_norms(const float* __restrict__ set1,
                             const float* __restrict__ set2,
                             float* __restrict__ sq1,
                             float* __restrict__ sq2) {
    int i = blockIdx.x * blockDim.x + threadIdx.x;
    const float* src;
    float* dst;
    if (i < BB * NN) {
        src = set1 + (size_t)i * CC; dst = sq1 + i;
    } else if (i < BB * (NN + MM)) {
        int j = i - BB * NN;
        src = set2 + (size_t)j * CC; dst = sq2 + j;
    } else {
        return;
    }
    float s = 0.f;
    for (int k = 0; k < CC; k += 4) {
        float4 v = *(const float4*)(src + k);
        s += v.x * v.x + v.y * v.y + v.z * v.z + v.w * v.w;
    }
    *dst = s;
}

// ---------------- fused GEMM (bf16 hi/lo split WMMA) + min-d^2 reductions ----------------
// grid = BB * (NN/128) blocks, 256 threads (8 waves). Each wave owns 16 set1 rows
// for ALL of M (row d^2-mins stay in registers); block loops M in 32-col tiles
// (two 16-col B tiles staged through LDS, two independent accumulators per wave).
// sqrt is deferred past the min reductions (monotonic).
__global__ __launch_bounds__(256) void hd_main(
        const float* __restrict__ set1, const float* __restrict__ set2,
        const float* __restrict__ sq1,  const float* __restrict__ sq2,
        float* __restrict__ rowMin, unsigned int* __restrict__ colBits) {

    // [hi/lo][tile u][kchunk][lane][16 bf16] = 16 KB
    __shared__ __attribute__((aligned(32))) unsigned short ldsB[2][2][4][32][16];
    __shared__ unsigned int ldsCol[32];

    const int t     = threadIdx.x;
    const int lane  = t & 31;
    const int wave  = t >> 5;
    const int lrow  = lane & 15;   // tile row (A) / tile col (B) owned by this lane
    const int khalf = lane >> 4;   // which K-half this lane holds (operand layout)

    const int b        = blockIdx.x >> 5;   // 32 row-strips per batch
    const int nStrip   = blockIdx.x & 31;
    const int tileBase = nStrip * 128 + wave * 16;

    // ---- A panel: 16 rows x 128 cols of set1, bf16 hi/lo, kept in VGPRs ----
    v16bf a_hi[4], a_lo[4];
    {
        const float* arow = set1 + ((size_t)b * NN + tileBase + lrow) * CC;
        for (int c = 0; c < 4; ++c) {
            for (int seg = 0; seg < 2; ++seg) {
                int k = c * 32 + seg * 16 + khalf * 8;
                float4 p0 = *(const float4*)(arow + k);
                float4 p1 = *(const float4*)(arow + k + 4);
                float f[8] = {p0.x, p0.y, p0.z, p0.w, p1.x, p1.y, p1.z, p1.w};
                for (int j = 0; j < 8; ++j) {
                    __bf16 hi = (__bf16)f[j];
                    a_hi[c][seg * 8 + j] = hi;
                    a_lo[c][seg * 8 + j] = (__bf16)(f[j] - (float)hi);
                }
            }
        }
    }

    // squared norms of the 8 set1 rows this lane's accumulators cover
    float s1v[8];
    for (int g = 0; g < 8; ++g)
        s1v[g] = sq1[(size_t)b * NN + tileBase + g + 8 * khalf];

    float rmin2[8];   // running min of d^2 per owned row
    for (int g = 0; g < 8; ++g) rmin2[g] = FLTMAX;

    for (int m0 = 0; m0 < MM; m0 += 32) {
        __syncthreads();   // protect LDS reuse vs previous iteration
        if (t < 32) ldsCol[t] = INF_BITS;

        // ---- stage set2 tiles (2 x 16 rows x 128 cols) into LDS, B-operand layout ----
        {
            int row = t >> 4;          // set2 row within a 16-row tile
            int kb  = (t & 15) * 8;    // 8 consecutive K per thread
            int c   = kb >> 5;
            int seg = (kb >> 4) & 1;
            int kh  = (kb >> 3) & 1;
            for (int u = 0; u < 2; ++u) {
                const float* brow =
                    set2 + ((size_t)b * MM + m0 + u * 16 + row) * CC + kb;
                float4 p0 = *(const float4*)(brow);
                float4 p1 = *(const float4*)(brow + 4);
                float f[8] = {p0.x, p0.y, p0.z, p0.w, p1.x, p1.y, p1.z, p1.w};
                unsigned int hw[4], lw[4];
                for (int j = 0; j < 4; ++j) {
                    __bf16 h0 = (__bf16)f[2 * j];
                    __bf16 h1 = (__bf16)f[2 * j + 1];
                    __bf16 l0 = (__bf16)(f[2 * j] - (float)h0);
                    __bf16 l1 = (__bf16)(f[2 * j + 1] - (float)h1);
                    hw[j] = (unsigned int)__builtin_bit_cast(unsigned short, h0) |
                            ((unsigned int)__builtin_bit_cast(unsigned short, h1) << 16);
                    lw[j] = (unsigned int)__builtin_bit_cast(unsigned short, l0) |
                            ((unsigned int)__builtin_bit_cast(unsigned short, l1) << 16);
                }
                uint4 phi = {hw[0], hw[1], hw[2], hw[3]};
                uint4 plo = {lw[0], lw[1], lw[2], lw[3]};
                *(uint4*)&ldsB[0][u][c][row + 16 * kh][seg * 8] = phi;
                *(uint4*)&ldsB[1][u][c][row + 16 * kh][seg * 8] = plo;
            }
        }
        __syncthreads();

        // ---- two 16x16 inner-product tiles: hi*hi + hi*lo + lo*hi over K=128 ----
        v8f acc0 = {0.f, 0.f, 0.f, 0.f, 0.f, 0.f, 0.f, 0.f};
        v8f acc1 = {0.f, 0.f, 0.f, 0.f, 0.f, 0.f, 0.f, 0.f};
        for (int c = 0; c < 4; ++c) {
            v16bf bhi0 = *(const v16bf*)&ldsB[0][0][c][lane][0];
            v16bf blo0 = *(const v16bf*)&ldsB[1][0][c][lane][0];
            v16bf bhi1 = *(const v16bf*)&ldsB[0][1][c][lane][0];
            v16bf blo1 = *(const v16bf*)&ldsB[1][1][c][lane][0];
            acc0 = __builtin_amdgcn_wmma_f32_16x16x32_bf16(false, a_hi[c], false, bhi0,
                                                           (short)0, acc0, false, false);
            acc1 = __builtin_amdgcn_wmma_f32_16x16x32_bf16(false, a_hi[c], false, bhi1,
                                                           (short)0, acc1, false, false);
            acc0 = __builtin_amdgcn_wmma_f32_16x16x32_bf16(false, a_hi[c], false, blo0,
                                                           (short)0, acc0, false, false);
            acc1 = __builtin_amdgcn_wmma_f32_16x16x32_bf16(false, a_hi[c], false, blo1,
                                                           (short)0, acc1, false, false);
            acc0 = __builtin_amdgcn_wmma_f32_16x16x32_bf16(false, a_lo[c], false, bhi0,
                                                           (short)0, acc0, false, false);
            acc1 = __builtin_amdgcn_wmma_f32_16x16x32_bf16(false, a_lo[c], false, bhi1,
                                                           (short)0, acc1, false, false);
        }

        // ---- epilogue on d^2 (sqrt deferred; clamp keeps values >= 0 for uint min) ----
        float s2_0 = sq2[(size_t)b * MM + m0 + lrow];
        float s2_1 = sq2[(size_t)b * MM + m0 + 16 + lrow];
        float cmin0 = FLTMAX, cmin1 = FLTMAX;
        for (int g = 0; g < 8; ++g) {
            float d0 = fmaxf(s1v[g] + s2_0 - 2.0f * acc0[g], 0.0f);
            float d1 = fmaxf(s1v[g] + s2_1 - 2.0f * acc1[g], 0.0f);
            rmin2[g] = fminf(rmin2[g], fminf(d0, d1));
            cmin0    = fminf(cmin0, d0);
            cmin1    = fminf(cmin1, d1);
        }
        // combine the two lane halves (rows g..g+7 vs g+8..g+15 share columns)
        cmin0 = fminf(cmin0, __shfl_xor(cmin0, 16, 32));
        cmin1 = fminf(cmin1, __shfl_xor(cmin1, 16, 32));
        if (lane < 16) {
            atomicMin(&ldsCol[lane],      __float_as_uint(cmin0));
            atomicMin(&ldsCol[lane + 16], __float_as_uint(cmin1));
        }
        __syncthreads();
        if (t < 32) atomicMin(&colBits[(size_t)b * MM + m0 + t], ldsCol[t]);
    }

    // ---- finalize row mins: lane-reduce d^2, take sqrt once per row ----
    for (int g = 0; g < 8; ++g) {
        float r = rmin2[g];
        r = fminf(r, __shfl_xor(r, 1, 32));
        r = fminf(r, __shfl_xor(r, 2, 32));
        r = fminf(r, __shfl_xor(r, 4, 32));
        r = fminf(r, __shfl_xor(r, 8, 32));
        if (lrow == 0)
            rowMin[(size_t)b * NN + tileBase + g + 8 * khalf] = sqrtf(r);
    }
}

// ---------------- per-batch means (col mins get their sqrt here) ----------------
__global__ void hd_finalize(const float* __restrict__ rowMin,
                            const unsigned int* __restrict__ colBits,
                            float* __restrict__ out) {
    __shared__ float red[256];
    int b = blockIdx.x, t = threadIdx.x;
    float sR = 0.f, sC = 0.f;
    for (int i = t; i < NN; i += 256) sR += rowMin[(size_t)b * NN + i];
    for (int i = t; i < MM; i += 256)
        sC += sqrtf(__uint_as_float(colBits[(size_t)b * MM + i]));
    red[t] = sR / (float)NN + sC / (float)MM;
    __syncthreads();
    for (int s = 128; s > 0; s >>= 1) {
        if (t < s) red[t] += red[t + s];
        __syncthreads();
    }
    if (t == 0) out[b] = red[0];
}

extern "C" void kernel_launch(void* const* d_in, const int* in_sizes, int n_in,
                              void* d_out, int out_size, void* d_ws, size_t ws_size,
                              hipStream_t stream) {
    (void)in_sizes; (void)n_in; (void)out_size; (void)ws_size;
    const float* set1 = (const float*)d_in[0];
    const float* set2 = (const float*)d_in[1];
    float* out = (float*)d_out;

    float* sq1            = (float*)d_ws;
    float* sq2            = sq1 + BB * NN;
    float* rowMin         = sq2 + BB * MM;
    unsigned int* colBits = (unsigned int*)(rowMin + BB * NN);

    hd_init_col<<<(BB * MM + 255) / 256, 256, 0, stream>>>(colBits);
    hd_row_norms<<<(BB * (NN + MM) + 255) / 256, 256, 0, stream>>>(set1, set2, sq1, sq2);
    hd_main<<<BB * (NN / 128), 256, 0, stream>>>(set1, set2, sq1, sq2, rowMin, colBits);
    hd_finalize<<<BB, 256, 0, stream>>>(rowMin, colBits, out);
}
